// DGCNN_15307263443061
// MI455X (gfx1250) — matrix operand order repaired and checked by reading, not compiled, not run on hardware
//
#include <hip/hip_runtime.h>
#include <hip/hip_bf16.h>
#include <math.h>

// ---------------- problem constants (match reference) ----------------
#define NN   100000
#define EE   1600000
#define BB   2000
#define NPG  50
#define KK   30
#define FF   128
#define HH   32
#define TD   97          // 3*H + 1
#define C1O  16          // conv1 out channels
#define C1K  15          // K/2 after maxpool
#define C2O  32
#define C2L  11          // 15-5+1
#define DENSE (C2O*C2L)  // 352

typedef __attribute__((ext_vector_type(16))) _Float16 v16h;
typedef __attribute__((ext_vector_type(8)))  float    v8f;

// ---------------- small utility kernels ----------------
__global__ void k_fill(float* p, float v, int n) {
    int t = blockIdx.x * blockDim.x + threadIdx.x;
    if (t < n) p[t] = v;
}

__global__ void k_count_deg(const int* __restrict__ dst, float* __restrict__ deg, int e) {
    int t = blockIdx.x * blockDim.x + threadIdx.x;
    if (t < e) atomicAdd(&deg[dst[t]], 1.0f);
}

__global__ void k_rsqrt_inplace(float* p, int n) {
    int t = blockIdx.x * blockDim.x + threadIdx.x;
    if (t < n) p[t] = rsqrtf(p[t]);
}

// ---------------- GEMM: Hout[M,32] = X[M,Kdim] @ W[Kdim,32] via WMMA f16 ----------------
// One wave -> one 16-row tile, two 16x16 accumulators (full 32 output cols).
__global__ void k_gemm_h32_wmma(const float* __restrict__ X, const float* __restrict__ W,
                                float* __restrict__ Hout, int M, int Kdim) {
    int wave = (blockIdx.x * blockDim.x + threadIdx.x) >> 5;
    int lane = threadIdx.x & 31;
    int row0 = wave * 16;
    if (row0 >= M) return;

    const int Ma  = lane & 15;             // A-fragment row
    const int kbA = (lane < 16) ? 0 : 8;   // A-fragment K base / 2 (k = 16h + e, contiguous)
    const int nB  = lane & 15;             // B/C column
    const int kbB = (lane < 16) ? 0 : 16;  // B-fragment K base (k = 16h + e, contiguous, matches A)

    v8f c0 = {}; v8f c1 = {};
    for (int kc = 0; kc < Kdim; kc += 32) {
        const float* xrow = X + (size_t)(row0 + Ma) * Kdim + kc;
        v16h a, b0, b1;
        #pragma unroll
        for (int e = 0; e < 16; ++e) {
            int v = e >> 1, lo = e & 1;
            int k = 2 * kbA + e; (void)v; (void)lo;
            a[e] = (_Float16)xrow[k];
        }
        #pragma unroll
        for (int e = 0; e < 16; ++e) {
            int k = kc + kbB + e;
            b0[e] = (_Float16)W[k * HH + nB];
            b1[e] = (_Float16)W[k * HH + 16 + nB];
        }
        c0 = __builtin_amdgcn_wmma_f32_16x16x32_f16(false, a, false, b0, (short)0, c0, false, false);
        c1 = __builtin_amdgcn_wmma_f32_16x16x32_f16(false, a, false, b1, (short)0, c1, false, false);
    }
    // f32 C 16x16 layout: VGPR r -> M = r + 8*(lane>=16), N = lane&15
    int rowAdd = (lane < 16) ? 0 : 8;
    #pragma unroll
    for (int r = 0; r < 8; ++r) {
        int m = row0 + r + rowAdd;
        Hout[(size_t)m * HH + nB]      = c0[r];
        Hout[(size_t)m * HH + 16 + nB] = c1[r];
    }
}

// ---------------- edge scatter: agg[dst] += h[src] * dinv[src]*dinv[dst] ----------------
// 8 lanes per edge, float4 per lane: coalesced 128B gather + 4-wide atomics (L2-resident).
__global__ void k_scatter32(const int* __restrict__ src, const int* __restrict__ dst,
                            const float* __restrict__ dinv, const float* __restrict__ h,
                            float* __restrict__ agg, int e) {
    int t  = blockIdx.x * blockDim.x + threadIdx.x;
    int ed = t >> 3;
    if (ed >= e) return;
    int c = (t & 7) * 4;
    int s = src[ed], d = dst[ed];
    float coef = dinv[s] * dinv[d];
    const float4 hv = *reinterpret_cast<const float4*>(h + (size_t)s * HH + c);
    float* ap = agg + (size_t)d * HH + c;
    atomicAdd(ap + 0, hv.x * coef);
    atomicAdd(ap + 1, hv.y * coef);
    atomicAdd(ap + 2, hv.z * coef);
    atomicAdd(ap + 3, hv.w * coef);
}

__global__ void k_finalize32(const float* __restrict__ agg, const float* __restrict__ h,
                             const float* __restrict__ dinv, const float* __restrict__ b,
                             float* __restrict__ xout, int n) { // n = N*32
    int t = blockIdx.x * blockDim.x + threadIdx.x;
    if (t >= n) return;
    int node = t >> 5, c = t & 31;
    float di = dinv[node];
    xout[t] = tanhf(agg[t] + h[t] * di * di + b[c]);
}

// ---------------- layer 4 (H -> 1) ----------------
__global__ void k_gemv_h1(const float* __restrict__ x3, const float* __restrict__ W4,
                          float* __restrict__ h4, int n) {
    int t = blockIdx.x * blockDim.x + threadIdx.x;
    if (t >= n) return;
    float acc = 0.f;
    #pragma unroll
    for (int c = 0; c < HH; ++c) acc += x3[(size_t)t * HH + c] * W4[c];
    h4[t] = acc;
}

__global__ void k_scatter1(const int* __restrict__ src, const int* __restrict__ dst,
                           const float* __restrict__ dinv, const float* __restrict__ h4,
                           float* __restrict__ agg4, int e) {
    int t = blockIdx.x * blockDim.x + threadIdx.x;
    if (t >= e) return;
    int s = src[t], d = dst[t];
    atomicAdd(&agg4[d], h4[s] * dinv[s] * dinv[d]);
}

__global__ void k_final1(const float* __restrict__ agg4, const float* __restrict__ h4,
                         const float* __restrict__ dinv, const float* __restrict__ b4,
                         float* __restrict__ x4, int n) {
    int t = blockIdx.x * blockDim.x + threadIdx.x;
    if (t >= n) return;
    float di = dinv[t];
    x4[t] = tanhf(agg4[t] + h4[t] * di * di + b4[0]);
}

// ---------------- sort-pool: stable descending rank on x4, gather top-K rows of [x1|x2|x3|x4] ----------------
__global__ void k_sortpool(const float* __restrict__ x1, const float* __restrict__ x2,
                           const float* __restrict__ x3, const float* __restrict__ x4,
                           float* __restrict__ pooled) {
    int g = blockIdx.x;
    __shared__ float v[NPG];
    __shared__ int order[KK];
    int tid = threadIdx.x; // 128
    if (tid < NPG) v[tid] = x4[g * NPG + tid];
    __syncthreads();
    if (tid < NPG) {
        float vi = v[tid];
        int r = 0;
        for (int j = 0; j < NPG; ++j) {
            float vj = v[j];
            r += (vj > vi) || (vj == vi && j < tid);  // stable argsort(-v)
        }
        if (r < KK) order[r] = tid;
    }
    __syncthreads();
    for (int idx = tid; idx < KK * TD; idx += blockDim.x) {
        int r = idx / TD, t = idx - r * TD;
        int n = g * NPG + order[r];
        float val = (t < 32) ? x1[(size_t)n * HH + t]
                  : (t < 64) ? x2[(size_t)n * HH + (t - 32)]
                  : (t < 96) ? x3[(size_t)n * HH + (t - 64)]
                             : x4[n];
        pooled[(size_t)g * KK * TD + idx] = val;
    }
}

// ---------------- conv1 (1->16, kernel TD, stride TD) + relu + maxpool(2,2) ----------------
__global__ void k_conv1pool(const float* __restrict__ pooled, const float* __restrict__ cw1,
                            const float* __restrict__ cb1, float* __restrict__ c1out) {
    int g = blockIdx.x;
    __shared__ float sp[KK * TD];
    __shared__ float sw[C1O * TD];
    int tid = threadIdx.x; // 256
    for (int i = tid; i < KK * TD; i += 256) sp[i] = pooled[(size_t)g * KK * TD + i];
    for (int i = tid; i < C1O * TD; i += 256) sw[i] = cw1[i];
    __syncthreads();
    if (tid < C1O * C1K) {
        int o = tid / C1K, j = tid % C1K;
        float m = -1e30f;
        for (int p = 0; p < 2; ++p) {
            int k = 2 * j + p;
            float acc = cb1[o];
            for (int t = 0; t < TD; ++t) acc += sp[k * TD + t] * sw[o * TD + t];
            acc = fmaxf(acc, 0.f);
            m = fmaxf(m, acc);
        }
        c1out[(size_t)g * (C1O * C1K) + o * C1K + j] = m;
    }
}

// ---------------- conv2 (16->32, kernel 5) + relu, flattened [B,352] ----------------
__global__ void k_conv2(const float* __restrict__ c1, const float* __restrict__ cw2,
                        const float* __restrict__ cb2, float* __restrict__ c2) {
    int g = blockIdx.x;
    int tid = threadIdx.x; // 384
    __shared__ float s1[C1O * C1K];
    for (int i = tid; i < C1O * C1K; i += blockDim.x) s1[i] = c1[(size_t)g * (C1O * C1K) + i];
    __syncthreads();
    if (tid < C2O * C2L) {
        int o = tid / C2L, j = tid % C2L;
        float acc = cb2[o];
        #pragma unroll
        for (int i = 0; i < C1O; ++i)
            #pragma unroll
            for (int w = 0; w < 5; ++w)
                acc += s1[i * C1K + j + w] * cw2[o * (C1O * 5) + i * 5 + w];
        c2[(size_t)g * DENSE + tid] = fmaxf(acc, 0.f);
    }
}

// ---------------- FC1 (352 -> 128) + relu ----------------
__global__ void k_fc1(const float* __restrict__ c2, const float* __restrict__ fw1,
                      const float* __restrict__ fb1, float* __restrict__ f1) {
    int g = blockIdx.x;
    int tid = threadIdx.x; // 128
    __shared__ float s2[DENSE];
    for (int i = tid; i < DENSE; i += 128) s2[i] = c2[(size_t)g * DENSE + i];
    __syncthreads();
    float acc = fb1[tid];
    for (int k = 0; k < DENSE; ++k) acc += s2[k] * fw1[(size_t)k * 128 + tid];
    f1[(size_t)g * 128 + tid] = fmaxf(acc, 0.f);
}

// ---------------- FC2 (128 -> 2) + log_softmax ----------------
__global__ void k_fc2_lsm(const float* __restrict__ f1, const float* __restrict__ fw2,
                          const float* __restrict__ fb2, float* __restrict__ out, int b) {
    int g = blockIdx.x * blockDim.x + threadIdx.x;
    if (g >= b) return;
    float z0 = fb2[0], z1 = fb2[1];
    for (int k = 0; k < 128; ++k) {
        float v = f1[(size_t)g * 128 + k];
        z0 += v * fw2[k * 2 + 0];
        z1 += v * fw2[k * 2 + 1];
    }
    float m = fmaxf(z0, z1);
    float lse = m + logf(__expf(z0 - m) + __expf(z1 - m));
    out[g * 2 + 0] = z0 - lse;
    out[g * 2 + 1] = z1 - lse;
}

// ---------------- host launch ----------------
extern "C" void kernel_launch(void* const* d_in, const int* in_sizes, int n_in,
                              void* d_out, int out_size, void* d_ws, size_t ws_size,
                              hipStream_t stream) {
    (void)in_sizes; (void)n_in; (void)out_size; (void)ws_size;

    const float* x   = (const float*)d_in[0];
    const int*   ei  = (const int*)d_in[1];
    const int*   src = ei;
    const int*   dst = ei + EE;
    // d_in[2] = batch (implicit: node/NPG)
    const float* W1 = (const float*)d_in[3];  const float* b1 = (const float*)d_in[4];
    const float* W2 = (const float*)d_in[5];  const float* b2 = (const float*)d_in[6];
    const float* W3 = (const float*)d_in[7];  const float* b3 = (const float*)d_in[8];
    const float* W4 = (const float*)d_in[9];  const float* b4 = (const float*)d_in[10];
    const float* cw1 = (const float*)d_in[11]; const float* cb1 = (const float*)d_in[12];
    const float* cw2 = (const float*)d_in[13]; const float* cb2 = (const float*)d_in[14];
    const float* fw1 = (const float*)d_in[15]; const float* fb1 = (const float*)d_in[16];
    const float* fw2 = (const float*)d_in[17]; const float* fb2 = (const float*)d_in[18];
    float* out = (float*)d_out;

    // workspace carve-up (256B aligned)
    char* ws = (char*)d_ws;
    size_t off = 0;
    auto carve = [&](size_t bytes) -> float* {
        float* p = (float*)(ws + off);
        off = (off + bytes + 255) & ~(size_t)255;
        return p;
    };
    float* dinv  = carve((size_t)NN * 4);
    float* h     = carve((size_t)NN * HH * 4);
    float* agg   = carve((size_t)NN * HH * 4);
    float* x1    = carve((size_t)NN * HH * 4);
    float* x2    = carve((size_t)NN * HH * 4);
    float* x3    = carve((size_t)NN * HH * 4);
    float* h4    = carve((size_t)NN * 4);
    float* agg4  = carve((size_t)NN * 4);
    float* x4    = carve((size_t)NN * 4);
    float* pooled= carve((size_t)BB * KK * TD * 4);
    float* c1    = carve((size_t)BB * C1O * C1K * 4);
    float* c2    = carve((size_t)BB * DENSE * 4);
    float* f1    = carve((size_t)BB * 128 * 4);

    const int T = 256;
    dim3 blk(T);
    int gN   = (NN + T - 1) / T;
    int gE   = (EE + T - 1) / T;
    int gNH  = (NN * HH + T - 1) / T;
    int gE8  = (EE * 8 + T - 1) / T;
    int gGemm = ((NN / 16) + 7) / 8;   // 8 waves/block, 1 tile/wave

    // degrees (self-loop = +1) -> dinv
    k_fill<<<gN, blk, 0, stream>>>(dinv, 1.0f, NN);
    k_count_deg<<<gE, blk, 0, stream>>>(dst, dinv, EE);
    k_rsqrt_inplace<<<gN, blk, 0, stream>>>(dinv, NN);

    // GCN layer 1
    k_gemm_h32_wmma<<<gGemm, blk, 0, stream>>>(x, W1, h, NN, FF);
    k_fill<<<gNH, blk, 0, stream>>>(agg, 0.0f, NN * HH);
    k_scatter32<<<gE8, blk, 0, stream>>>(src, dst, dinv, h, agg, EE);
    k_finalize32<<<gNH, blk, 0, stream>>>(agg, h, dinv, b1, x1, NN * HH);

    // GCN layer 2
    k_gemm_h32_wmma<<<gGemm, blk, 0, stream>>>(x1, W2, h, NN, HH);
    k_fill<<<gNH, blk, 0, stream>>>(agg, 0.0f, NN * HH);
    k_scatter32<<<gE8, blk, 0, stream>>>(src, dst, dinv, h, agg, EE);
    k_finalize32<<<gNH, blk, 0, stream>>>(agg, h, dinv, b2, x2, NN * HH);

    // GCN layer 3
    k_gemm_h32_wmma<<<gGemm, blk, 0, stream>>>(x2, W3, h, NN, HH);
    k_fill<<<gNH, blk, 0, stream>>>(agg, 0.0f, NN * HH);
    k_scatter32<<<gE8, blk, 0, stream>>>(src, dst, dinv, h, agg, EE);
    k_finalize32<<<gNH, blk, 0, stream>>>(agg, h, dinv, b3, x3, NN * HH);

    // GCN layer 4 (H -> 1)
    k_gemv_h1<<<gN, blk, 0, stream>>>(x3, W4, h4, NN);
    k_fill<<<gN, blk, 0, stream>>>(agg4, 0.0f, NN);
    k_scatter1<<<gE, blk, 0, stream>>>(src, dst, dinv, h4, agg4, EE);
    k_final1<<<gN, blk, 0, stream>>>(agg4, h4, dinv, b4, x4, NN);

    // sort-pool -> pooled [B, K, TD]
    k_sortpool<<<BB, dim3(128), 0, stream>>>(x1, x2, x3, x4, pooled);

    // conv1 + relu + maxpool -> [B,16,15]
    k_conv1pool<<<BB, dim3(256), 0, stream>>>(pooled, cw1, cb1, c1);

    // conv2 + relu -> [B,352]
    k_conv2<<<BB, dim3(384), 0, stream>>>(c1, cw2, cb2, c2);

    // FC1 + relu -> [B,128]
    k_fc1<<<BB, dim3(128), 0, stream>>>(c2, fw1, fb1, f1);

    // FC2 + log_softmax -> [B,2]
    k_fc2_lsm<<<(BB + T - 1) / T, blk, 0, stream>>>(f1, fw2, fb2, out, BB);
}
